// CudaSafeLinear_29377576304853
// MI455X (gfx1250) — matrix operand order repaired and checked by reading, not compiled, not run on hardware
//
#include <hip/hip_runtime.h>

// ---- CDNA5 WMMA vector types -------------------------------------------------
typedef __attribute__((ext_vector_type(16))) __bf16 v16bf;
typedef __attribute__((ext_vector_type(8)))  __bf16 v8bf;
typedef __attribute__((ext_vector_type(8)))  float  v8f;
typedef __attribute__((ext_vector_type(4)))  int    v4i;

// Typed pointers for the async global->LDS DMA builtin (param type is int4*)
typedef __attribute__((address_space(1))) v4i v4i_g;   // global
typedef __attribute__((address_space(3))) v4i v4i_l;   // LDS

#if __has_builtin(__builtin_amdgcn_global_load_async_to_lds_b128)
#define USE_ASYNC_LDS 1
#endif

__device__ __forceinline__ void wait_async0() {
#if __has_builtin(__builtin_amdgcn_s_wait_asynccnt)
  __builtin_amdgcn_s_wait_asynccnt(0);
#else
  asm volatile("s_wait_asynccnt 0x0" ::: "memory");
#endif
}

// Problem dims (fixed by reference)
#define M_DIM 8192
#define N_DIM 4096
#define K_DIM 4096

constexpr int BM = 128;            // block tile M
constexpr int BN = 128;            // block tile N

// fp32 -> bf16 (round-to-nearest, ties up) packed pairwise: 2 adds + 1 v_perm_b32
__device__ __forceinline__ unsigned int pk_bf16(float a, float b) {
  unsigned int ua = __float_as_uint(a) + 0x8000u;
  unsigned int ub = __float_as_uint(b) + 0x8000u;
  return __builtin_amdgcn_perm(ub, ua, 0x07060302u);
}

// =============================================================================
// Pre-pass: fp32 -> bf16, 8 floats / thread, pure bandwidth
// =============================================================================
__global__ __launch_bounds__(256)
void cvt_bf16_kernel(const float* __restrict__ src, unsigned int* __restrict__ dst) {
  const size_t i = ((size_t)blockIdx.x * 256 + threadIdx.x) * 8;
  float4 a = *(const float4*)(src + i);
  float4 b = *(const float4*)(src + i + 4);
  uint4 o;
  o.x = pk_bf16(a.x, a.y);
  o.y = pk_bf16(a.z, a.w);
  o.z = pk_bf16(b.x, b.y);
  o.w = pk_bf16(b.z, b.w);
  *(uint4*)(dst + i / 2) = o;
}

// =============================================================================
// Main GEMM: bf16 inputs from d_ws, BK=64, double-buffered LDS, 16 WMMA / tile
// =============================================================================
constexpr int TK  = 64;            // K per LDS tile
constexpr int LDB = TK + 8;        // 72 bf16 (144B = 36 banks) conflict-free
constexpr int KT2 = K_DIM / TK;    // 64 iterations

__global__ __launch_bounds__(256)
void gemm_bf16(const __bf16* __restrict__ Xb, const __bf16* __restrict__ Wb,
               const float* __restrict__ bias, float* __restrict__ out) {
  __shared__ alignas(16) __bf16 As[2][BM * LDB];   // 2 x 18432 B
  __shared__ alignas(16) __bf16 Bs[2][BN * LDB];   // 2 x 18432 B

  const int tid  = threadIdx.x;
  const int lane = tid & 31;
  const int wave = tid >> 5;
  const int wr   = wave >> 2;      // 0..1 : 64-row half
  const int wc   = wave & 3;       // 0..3 : 32-col group
  const int l16  = lane & 15;
  const int hs   = lane >> 4;

  const int m0 = blockIdx.y * BM;
  const int n0 = blockIdx.x * BN;

  // staging: 128 rows x 64 bf16 per matrix; 16B chunks, 4 passes of 32 rows
  const int srow = tid >> 3;       // 0..31
  const int schk = tid & 7;        // 8-bf16 chunk
  const __bf16* Ag = Xb + (size_t)(m0 + srow) * K_DIM + schk * 8;
  const __bf16* Bg = Wb + (size_t)(n0 + srow) * K_DIM + schk * 8;

#ifdef USE_ASYNC_LDS
  // DMA global -> LDS directly (ASYNCcnt), no VGPR round-trip, no ds_store
  auto stage = [&](int buf, int k) {
    #pragma unroll
    for (int it = 0; it < 4; ++it) {
      __builtin_amdgcn_global_load_async_to_lds_b128(
          (v4i_g*)(Ag + (size_t)(32 * it) * K_DIM + k),
          (v4i_l*)&As[buf][(srow + 32 * it) * LDB + schk * 8], 0, 0);
      __builtin_amdgcn_global_load_async_to_lds_b128(
          (v4i_g*)(Bg + (size_t)(32 * it) * K_DIM + k),
          (v4i_l*)&Bs[buf][(srow + 32 * it) * LDB + schk * 8], 0, 0);
    }
  };
#else
  uint4 pa[4], pb[4];
  auto fetch = [&](int k) {
    #pragma unroll
    for (int it = 0; it < 4; ++it) {
      pa[it] = *(const uint4*)(Ag + (size_t)(32 * it) * K_DIM + k);
      pb[it] = *(const uint4*)(Bg + (size_t)(32 * it) * K_DIM + k);
    }
  };
  auto commit = [&](int buf) {
    #pragma unroll
    for (int it = 0; it < 4; ++it) {
      *(uint4*)&As[buf][(srow + 32 * it) * LDB + schk * 8] = pa[it];
      *(uint4*)&Bs[buf][(srow + 32 * it) * LDB + schk * 8] = pb[it];
    }
  };
#endif

  // accumulators seeded with bias (C/D layout: N = lane&15, M = v + 8*hs)
  float bv[2];
  #pragma unroll
  for (int nb = 0; nb < 2; ++nb)
    bv[nb] = bias[n0 + wc * 32 + nb * 16 + l16];

  v8f acc[4][2];
  #pragma unroll
  for (int mb = 0; mb < 4; ++mb)
    #pragma unroll
    for (int nb = 0; nb < 2; ++nb)
      #pragma unroll
      for (int i = 0; i < 8; ++i)
        acc[mb][nb][i] = bv[nb];

#ifdef USE_ASYNC_LDS
  stage(0, 0);
  wait_async0();
  __syncthreads();
#else
  fetch(0);
  commit(0);
  __syncthreads();
#endif

  for (int kt = 0; kt < KT2; ++kt) {
    const int cur = kt & 1;
#ifdef USE_ASYNC_LDS
    if (kt + 1 < KT2) stage(cur ^ 1, (kt + 1) * TK);  // DMA next tile under WMMA
#else
    if (kt + 1 < KT2) fetch((kt + 1) * TK);
#endif

    #pragma unroll
    for (int kk = 0; kk < 2; ++kk) {          // two K=32 slabs per LDS tile
      v16bf afr[4];
      #pragma unroll
      for (int mb = 0; mb < 4; ++mb) {
        const __bf16* p = &As[cur][(wr * 64 + mb * 16 + l16) * LDB + kk * 32 + hs * 8];
        union { v16bf v; v8bf h[2]; } u;
        u.h[0] = *(const v8bf*)p;             // ds_load_b128
        u.h[1] = *(const v8bf*)(p + 16);
        afr[mb] = u.v;
      }
      v16bf bfr[2];
      #pragma unroll
      for (int nb = 0; nb < 2; ++nb) {
        const __bf16* p = &Bs[cur][(wc * 32 + nb * 16 + l16) * LDB + kk * 32 + hs * 8];
        union { v16bf v; v8bf h[2]; } u;
        u.h[0] = *(const v8bf*)p;
        u.h[1] = *(const v8bf*)(p + 16);
        bfr[nb] = u.v;
      }
      #pragma unroll
      for (int mb = 0; mb < 4; ++mb)
        #pragma unroll
        for (int nb = 0; nb < 2; ++nb)
          acc[mb][nb] = __builtin_amdgcn_wmma_f32_16x16x32_bf16(
              false, afr[mb], false, bfr[nb], (short)0, acc[mb][nb], false, false);
    }

#ifdef USE_ASYNC_LDS
    wait_async0();                            // next-tile DMA complete
#else
    if (kt + 1 < KT2) commit(cur ^ 1);
#endif
    __syncthreads();
  }

  #pragma unroll
  for (int mb = 0; mb < 4; ++mb) {
    const int row = m0 + wr * 64 + mb * 16 + hs * 8;
    #pragma unroll
    for (int nb = 0; nb < 2; ++nb) {
      const int col = n0 + wc * 32 + nb * 16 + l16;
      float* op = out + (size_t)row * N_DIM + col;
      #pragma unroll
      for (int r = 0; r < 8; ++r)
        op[(size_t)r * N_DIM] = acc[mb][nb][r];
    }
  }
}

// =============================================================================
// Fallback: fused fp32->bf16 conversion in the GEMM (used only if d_ws too small)
// =============================================================================
constexpr int FK  = 32;
constexpr int FLD = FK + 8;        // 40
constexpr int FKT = K_DIM / FK;

__global__ __launch_bounds__(256)
void gemm_fused(const float* __restrict__ X, const float* __restrict__ W,
                const float* __restrict__ bias, float* __restrict__ out) {
  __shared__ alignas(16) __bf16 As[2][BM * FLD];
  __shared__ alignas(16) __bf16 Bs[2][BN * FLD];

  const int tid  = threadIdx.x;
  const int lane = tid & 31;
  const int wave = tid >> 5;
  const int wr   = wave >> 2;
  const int wc   = wave & 3;
  const int l16  = lane & 15;
  const int hs   = lane >> 4;

  const int m0 = blockIdx.y * BM;
  const int n0 = blockIdx.x * BN;

  const int srow = tid >> 3;
  const int scol = tid & 7;
  const float* Ag = X + (size_t)(m0 + srow) * K_DIM + scol * 4;
  const float* Bg = W + (size_t)(n0 + srow) * K_DIM + scol * 4;

  float4 pa[4], pb[4];
  auto fetch = [&](int k) {
    #pragma unroll
    for (int it = 0; it < 4; ++it) {
      pa[it] = *(const float4*)(Ag + (size_t)(32 * it) * K_DIM + k);
      pb[it] = *(const float4*)(Bg + (size_t)(32 * it) * K_DIM + k);
    }
  };
  auto commit = [&](int buf) {
    #pragma unroll
    for (int it = 0; it < 4; ++it) {
      uint2 va, vb;
      va.x = pk_bf16(pa[it].x, pa[it].y);
      va.y = pk_bf16(pa[it].z, pa[it].w);
      vb.x = pk_bf16(pb[it].x, pb[it].y);
      vb.y = pk_bf16(pb[it].z, pb[it].w);
      *(uint2*)&As[buf][(srow + 32 * it) * FLD + scol * 4] = va;
      *(uint2*)&Bs[buf][(srow + 32 * it) * FLD + scol * 4] = vb;
    }
  };

  float bv[2];
  #pragma unroll
  for (int nb = 0; nb < 2; ++nb)
    bv[nb] = bias[n0 + wc * 32 + nb * 16 + l16];

  v8f acc[4][2];
  #pragma unroll
  for (int mb = 0; mb < 4; ++mb)
    #pragma unroll
    for (int nb = 0; nb < 2; ++nb)
      #pragma unroll
      for (int i = 0; i < 8; ++i)
        acc[mb][nb][i] = bv[nb];

  fetch(0);
  commit(0);
  __syncthreads();

  for (int kt = 0; kt < FKT; ++kt) {
    const int cur = kt & 1;
    if (kt + 1 < FKT) fetch((kt + 1) * FK);

    v16bf afr[4];
    #pragma unroll
    for (int mb = 0; mb < 4; ++mb) {
      const __bf16* p = &As[cur][(wr * 64 + mb * 16 + l16) * FLD + hs * 8];
      union { v16bf v; v8bf h[2]; } u;
      u.h[0] = *(const v8bf*)p;
      u.h[1] = *(const v8bf*)(p + 16);
      afr[mb] = u.v;
    }
    v16bf bfr[2];
    #pragma unroll
    for (int nb = 0; nb < 2; ++nb) {
      const __bf16* p = &Bs[cur][(wc * 32 + nb * 16 + l16) * FLD + hs * 8];
      union { v16bf v; v8bf h[2]; } u;
      u.h[0] = *(const v8bf*)p;
      u.h[1] = *(const v8bf*)(p + 16);
      bfr[nb] = u.v;
    }
    #pragma unroll
    for (int mb = 0; mb < 4; ++mb)
      #pragma unroll
      for (int nb = 0; nb < 2; ++nb)
        acc[mb][nb] = __builtin_amdgcn_wmma_f32_16x16x32_bf16(
            false, afr[mb], false, bfr[nb], (short)0, acc[mb][nb], false, false);

    if (kt + 1 < FKT) commit(cur ^ 1);
    __syncthreads();
  }

  #pragma unroll
  for (int mb = 0; mb < 4; ++mb) {
    const int row = m0 + wr * 64 + mb * 16 + hs * 8;
    #pragma unroll
    for (int nb = 0; nb < 2; ++nb) {
      const int col = n0 + wc * 32 + nb * 16 + l16;
      float* op = out + (size_t)row * N_DIM + col;
      #pragma unroll
      for (int r = 0; r < 8; ++r)
        op[(size_t)r * N_DIM] = acc[mb][nb][r];
    }
  }
}

// =============================================================================
extern "C" void kernel_launch(void* const* d_in, const int* in_sizes, int n_in,
                              void* d_out, int out_size, void* d_ws, size_t ws_size,
                              hipStream_t stream) {
  const float* x    = (const float*)d_in[0];
  const float* w    = (const float*)d_in[1];
  const float* bias = (const float*)d_in[2];
  float* out        = (float*)d_out;

  const size_t xElems = (size_t)M_DIM * K_DIM;   // 33554432
  const size_t wElems = (size_t)N_DIM * K_DIM;   // 16777216
  const size_t need   = (xElems + wElems) * sizeof(__bf16);   // 96 MiB

  dim3 grid(N_DIM / BN, M_DIM / BM);             // 32 x 64

  if (ws_size >= need) {
    __bf16* Xb = (__bf16*)d_ws;
    __bf16* Wb = Xb + xElems;
    cvt_bf16_kernel<<<(unsigned)(xElems / 2048), 256, 0, stream>>>(x, (unsigned int*)Xb);
    cvt_bf16_kernel<<<(unsigned)(wElems / 2048), 256, 0, stream>>>(w, (unsigned int*)Wb);
    gemm_bf16<<<grid, dim3(256), 0, stream>>>(Xb, Wb, bias, out);
  } else {
    gemm_fused<<<grid, dim3(256), 0, stream>>>(x, w, bias, out);
  }
}
